// OptimizedBiLSTM_37160057045542
// MI455X (gfx1250) — matrix operand order repaired and checked by reading, not compiled, not run on hardware
//
#include <hip/hip_runtime.h>
#include <hip/hip_bf16.h>
#include <hip/hip_fp16.h>

#define B_    256
#define T_    512
#define E_    50
#define H_    50
#define G4    200   // 4*H
#define EPAD  64    // padded embedding row (halves)

typedef __attribute__((ext_vector_type(16))) _Float16 v16h;
typedef __attribute__((ext_vector_type(8)))  _Float16 h8;
typedef __attribute__((ext_vector_type(8)))  float    v8f;

// Branch-free fast activations: v_exp_f32 + v_rcp_f32, no divergent range checks.
__device__ __forceinline__ float sigf(float x) {
    return __builtin_amdgcn_rcpf(1.0f + __expf(-x));   // saturates cleanly at +/-inf
}
__device__ __forceinline__ float tanhfast(float x) {
    float a = __builtin_fabsf(x);
    float e = __expf(2.0f * a);                        // inf for large a -> r = 1
    float r = 1.0f - 2.0f * __builtin_amdgcn_rcpf(e + 1.0f);
    return __builtin_copysignf(r, x);
}

__device__ __forceinline__ v16h packa(h8 lo, h8 hi8) {
    v16h a;
#pragma unroll
    for (int e = 0; e < 8; ++e) { a[e] = lo[e]; a[e + 8] = hi8[e]; }
    return a;
}

// ---------------------------------------------------------------------------
// Kernel 1: embedding gather -> f16, padded to 64 halves per token (pad = 0)
// ---------------------------------------------------------------------------
__global__ void k_gather(const int* __restrict__ x, const float* __restrict__ tbl,
                         _Float16* __restrict__ emb16) {
    int gid   = blockIdx.x * blockDim.x + threadIdx.x;
    int token = gid >> 6;
    int col   = gid & 63;
    if (token >= B_ * T_) return;
    float v = 0.0f;
    if (col < E_) v = tbl[(long long)x[token] * E_ + col];
    emb16[(long long)token * EPAD + col] = (_Float16)v;
}

// ---------------------------------------------------------------------------
// Kernel 2: rep_mean (sum over ALL T / length) and rep_max (over ALL T), f32
// ---------------------------------------------------------------------------
__global__ void k_meanmax(const int* __restrict__ x, const int* __restrict__ lengths,
                          const float* __restrict__ tbl, float* __restrict__ rep) {
    int b = blockIdx.x;
    int e = threadIdx.x;
    if (e >= E_) return;
    float s = 0.0f, mx = -3.402823466e38f;
    for (int t = 0; t < T_; ++t) {
        float v = tbl[(long long)x[b * T_ + t] * E_ + e];
        s += v;
        mx = fmaxf(mx, v);
    }
    rep[b * G4 + 2 * H_ + e] = s / (float)lengths[b];
    rep[b * G4 + 3 * H_ + e] = mx;
}

// ---------------------------------------------------------------------------
// B-fragment builder for v_wmma_f32_16x16x32_f16.
// B layout (32x16, KxN): lane n = L&15; lanes 0-15 hold K=0..7,16..23, lanes 16-31
// hold K=8..15,24..31 (matches A operand order).
// Combined operand rows n of [W_ihT ; pad ; W_hhT ; pad]:
//   K in [0,50)   -> W_ih[n][K]
//   K in [64,114) -> W_hh[n][K-64]
//   else 0 (pad). n >= 200 -> all zero (dead tiles 13..15).
// ---------------------------------------------------------------------------
__device__ __forceinline__ v16h make_bfrag(const float* __restrict__ Wih,
                                           const float* __restrict__ Whh,
                                           int n, int q, int hi) {
    v16h bb;
#pragma unroll
    for (int e = 0; e < 16; ++e) {
        int k = 32 * q + (e < 8 ? 8 * hi + e : 16 + 8 * hi + (e - 8));
        float v = 0.0f;
        if (n < G4) {
            if (k < E_)                      v = Wih[n * E_ + k];
            else if (k >= 64 && k < 64 + H_) v = Whh[n * H_ + (k - 64)];
        }
        bb[e] = (_Float16)v;
    }
    return bb;
}

// ---------------------------------------------------------------------------
// Kernel 3: fused BiLSTM scan. One workgroup = 16 batch rows x 1 direction.
// Per step: G[16,200] = [emb_t | h] @ [W_ihT ; W_hhT]  (16 padded N-tiles x
// 4 K-chunks of v_wmma_f32_16x16x32_f16; every wave does 2 tiles, no EXEC
// games around WMMA). emb K-chunks come straight from global (b128, L2-hot,
// issued before the barrier); h K-chunks from LDS. Gate math is branch-free
// f32 (exp/rcp) with the cell state held in registers.
// ---------------------------------------------------------------------------
__global__ void __launch_bounds__(256, 1)
k_bilstm(const _Float16* __restrict__ emb16, const int* __restrict__ lengths,
         const float* __restrict__ Wih_f, const float* __restrict__ Whh_f,
         const float* __restrict__ bih_f, const float* __restrict__ bhh_f,
         const float* __restrict__ Wih_b, const float* __restrict__ Whh_b,
         const float* __restrict__ bih_b, const float* __restrict__ bhh_b,
         float* __restrict__ rep) {
    __shared__ _Float16 lds_h[16][64];     // h operand tile (cols 50..63 stay 0)
    __shared__ float    lds_g[16][256];    // gate pre-activations (cols 200+ dead)
    __shared__ float    lds_bias[G4];
    __shared__ int      lds_last[16];

    const int tid  = threadIdx.x;
    const int g    = blockIdx.x;
    const int dir  = g & 1;
    const int b0   = (g >> 1) * 16;
    const int lane = tid & 31;
    const int wv   = tid >> 5;      // wave id 0..7
    const int nlo  = lane & 15;
    const int hi   = lane >> 4;

    const float* Wih = dir ? Wih_b : Wih_f;
    const float* Whh = dir ? Whh_b : Whh_f;
    const float* bih = dir ? bih_b : bih_f;
    const float* bhh = dir ? bhh_b : bhh_f;

    // ---- init LDS: h = 0 (pads stay 0), bias, last index ----
    {
        uint* hw = (uint*)&lds_h[0][0];             // 16*64 halves = 512 words
        for (int i = tid; i < 512; i += 256) hw[i] = 0u;
        for (int i = tid; i < G4; i += 256) lds_bias[i] = bih[i] + bhh[i];
        if (tid < 16) {
            int L = lengths[b0 + tid];
            L = L < 1 ? 1 : (L > T_ ? T_ : L);
            lds_last[tid] = L - 1;
        }
    }

    // ---- loop-invariant B fragments: wave wv owns N-tiles {wv, wv+8} ----
    const int tn0 = wv, tn1 = wv + 8;
    v16h bf0[4], bf1[4];
#pragma unroll
    for (int q = 0; q < 4; ++q) {
        bf0[q] = make_bfrag(Wih, Whh, tn0 * 16 + nlo, q, hi);
        bf1[q] = make_bfrag(Wih, Whh, tn1 * 16 + nlo, q, hi);
    }

    __syncthreads();

    float cst[4] = {0.0f, 0.0f, 0.0f, 0.0f};   // cell state, 800 elems / 256 thr

    for (int t = 0; t < T_; ++t) {
        const int tt = dir ? (T_ - 1 - t) : t;

        // ---- emb A-chunks straight from global; independent of recurrence,
        //      issued before the barrier so they overlap the previous gates ----
        const _Float16* erow = emb16 + ((long long)(b0 + nlo) * T_ + tt) * EPAD;
        h8 e0a = *(const h8*)(erow + 8 * hi);
        h8 e0b = *(const h8*)(erow + 16 + 8 * hi);
        h8 e1a = *(const h8*)(erow + 32 + 8 * hi);
        h8 e1b = *(const h8*)(erow + 48 + 8 * hi);

        // ---- wave 0: prefetch next timestep's emb rows (global_prefetch) ----
        if (wv == 0 && t + 1 < T_) {
            const int tn = dir ? (T_ - 2 - t) : (t + 1);
            __builtin_prefetch((const void*)(emb16 + ((long long)(b0 + nlo) * T_ + tn) * EPAD),
                               0, 3);
        }

        __syncthreads();   // h(t-1) visible in lds_h

        // ---- WMMA: A layout lanes0-15 K=0..7/16..23, lanes16-31 +8 ----
        v8f acc0 = {};
        v8f acc1 = {};
        {
            v16h a = packa(e0a, e0b);                       // K 0..31  (emb)
            acc0 = __builtin_amdgcn_wmma_f32_16x16x32_f16(false, a, false, bf0[0], (short)0, acc0, false, false);
            acc1 = __builtin_amdgcn_wmma_f32_16x16x32_f16(false, a, false, bf1[0], (short)0, acc1, false, false);
        }
        {
            v16h a = packa(e1a, e1b);                       // K 32..63 (emb pad)
            acc0 = __builtin_amdgcn_wmma_f32_16x16x32_f16(false, a, false, bf0[1], (short)0, acc0, false, false);
            acc1 = __builtin_amdgcn_wmma_f32_16x16x32_f16(false, a, false, bf1[1], (short)0, acc1, false, false);
        }
#pragma unroll
        for (int q = 2; q < 4; ++q) {                       // K 64..127 (h)
            h8 hlo = *(const h8*)&lds_h[nlo][32 * (q - 2) + 8 * hi];
            h8 hhi = *(const h8*)&lds_h[nlo][32 * (q - 2) + 16 + 8 * hi];
            v16h a = packa(hlo, hhi);
            acc0 = __builtin_amdgcn_wmma_f32_16x16x32_f16(false, a, false, bf0[q], (short)0, acc0, false, false);
            acc1 = __builtin_amdgcn_wmma_f32_16x16x32_f16(false, a, false, bf1[q], (short)0, acc1, false, false);
        }

        // ---- scatter accumulators (C/D layout: vgpr r -> row r+8*hi) ----
        {
            int n = tn0 * 16 + nlo;
#pragma unroll
            for (int r = 0; r < 8; ++r) lds_g[r + 8 * hi][n] = acc0[r];
        }
        {
            int n = tn1 * 16 + nlo;                         // up to 255: pad cols
#pragma unroll
            for (int r = 0; r < 8; ++r) lds_g[r + 8 * hi][n] = acc1[r];
        }
        __syncthreads();

        // ---- gates: 16 rows x 50 hidden = 800 elements, branch-free math ----
#pragma unroll
        for (int r = 0; r < 4; ++r) {
            int idx = tid + 256 * r;
            if (idx < 16 * H_) {
                int m2 = idx / H_;
                int j  = idx % H_;
                float gi = lds_g[m2][j]          + lds_bias[j];
                float gf = lds_g[m2][H_ + j]     + lds_bias[H_ + j];
                float gg = lds_g[m2][2 * H_ + j] + lds_bias[2 * H_ + j];
                float go = lds_g[m2][3 * H_ + j] + lds_bias[3 * H_ + j];
                float c  = sigf(gf) * cst[r] + sigf(gi) * tanhfast(gg);
                float h  = sigf(go) * tanhfast(c);
                cst[r] = c;
                lds_h[m2][j] = (_Float16)h;
                if (tt == lds_last[m2])
                    rep[(b0 + m2) * G4 + dir * H_ + j] = h;
            }
        }
        __syncthreads();
    }
}

// ---------------------------------------------------------------------------
// Kernel 4: head — out[b] = rep[b] @ W_linT + b_lin  ([256,200]x[200,3])
// ---------------------------------------------------------------------------
__global__ void k_head(const float* __restrict__ rep, const float* __restrict__ Wlin,
                       const float* __restrict__ blin, float* __restrict__ out) {
    int b = threadIdx.x;
    if (b >= B_) return;
    float s0 = blin[0], s1 = blin[1], s2 = blin[2];
    for (int k = 0; k < G4; ++k) {
        float r = rep[b * G4 + k];
        s0 += r * Wlin[k];
        s1 += r * Wlin[G4 + k];
        s2 += r * Wlin[2 * G4 + k];
    }
    out[b * 3 + 0] = s0;
    out[b * 3 + 1] = s1;
    out[b * 3 + 2] = s2;
}

extern "C" void kernel_launch(void* const* d_in, const int* in_sizes, int n_in,
                              void* d_out, int out_size, void* d_ws, size_t ws_size,
                              hipStream_t stream) {
    const int*   x     = (const int*)d_in[0];
    const int*   len   = (const int*)d_in[1];
    // d_in[2] = bows (unused by reference)
    const float* tbl   = (const float*)d_in[3];
    const float* Wih_f = (const float*)d_in[4];
    const float* Whh_f = (const float*)d_in[5];
    const float* bih_f = (const float*)d_in[6];
    const float* bhh_f = (const float*)d_in[7];
    const float* Wih_b = (const float*)d_in[8];
    const float* Whh_b = (const float*)d_in[9];
    const float* bih_b = (const float*)d_in[10];
    const float* bhh_b = (const float*)d_in[11];
    const float* Wlin  = (const float*)d_in[12];
    const float* blin  = (const float*)d_in[13];
    float* out = (float*)d_out;

    _Float16* emb16 = (_Float16*)d_ws;
    float*    rep   = (float*)((char*)d_ws + (size_t)B_ * T_ * EPAD * sizeof(_Float16));

    k_gather <<<(B_ * T_ * EPAD) / 256, 256, 0, stream>>>(x, tbl, emb16);
    k_meanmax<<<B_, 64, 0, stream>>>(x, len, tbl, rep);
    k_bilstm <<<(B_ / 16) * 2, 256, 0, stream>>>(emb16, len,
                                                 Wih_f, Whh_f, bih_f, bhh_f,
                                                 Wih_b, Whh_b, bih_b, bhh_b, rep);
    k_head   <<<1, 256, 0, stream>>>(rep, Wlin, blin, out);
}
